// MultiHeadAttention_84550726189708
// MI455X (gfx1250) — matrix operand order repaired and checked by reading, not compiled, not run on hardware
//
#include <hip/hip_runtime.h>

// ---------------------------------------------------------------------------
// Graph-attention w/ hop+edge type embeddings for MI455X (gfx1250, wave32).
// All GEMMs on v_wmma_f32_16x16x32_f16. Operands are pre-converted to f16 in
// fragment-friendly (N-major) layouts so every WMMA B-fragment is exactly two
// b128 loads. Scores live in a 16x512 LDS strip; scatter-by-type re-aggregates
// with ds_add_f32 into [16x96] and folds back into the output via WMMA.
// ---------------------------------------------------------------------------

typedef __attribute__((ext_vector_type(16))) _Float16 v16h;
typedef __attribute__((ext_vector_type(8)))  _Float16 v8h;
typedef __attribute__((ext_vector_type(8)))  float    v8f;

#define BSZ   8
#define LSEQ  512
#define HIDN  768
#define NHEAD 12
#define DH    64
#define THOP  32
#define TEDGE 64

__device__ __forceinline__ v8f wmma16(v16h a, v16h b, v8f c) {
  return __builtin_amdgcn_wmma_f32_16x16x32_f16(
      false, a, false, b, (short)0, c, false, false);
}

// ---- A-fragment (16x32, MxK) from contiguous-K f16 memory/LDS.
// lane: m = lane&15, koff = (lane>>4)*8 ; elem i -> K = (i>=8?16:0)+koff+(i&7)
// -> two contiguous 8xf16 runs at +koff and +koff+16 : two b128 loads.
__device__ __forceinline__ v16h load_a_f16v(const _Float16* p0, int ld) {
  int lane = threadIdx.x & 31;
  int m = lane & 15, koff = (lane >> 4) * 8;
  const _Float16* p = p0 + (size_t)m * ld + koff;
  v8h lo = *(const v8h*)(p);
  v8h hi = *(const v8h*)(p + 16);
  return __builtin_shufflevector(lo, hi, 0, 1, 2, 3, 4, 5, 6, 7,
                                         8, 9, 10, 11, 12, 13, 14, 15);
}
// A-fragment from f32 LDS (scatter table): same K pattern, cvt on load.
__device__ __forceinline__ v16h load_a_f32(const float* p0, int ld) {
  int lane = threadIdx.x & 31;
  int m = lane & 15, koff = (lane >> 4) * 8;
  const float* p = p0 + (size_t)m * ld + koff;
  v16h a;
#pragma unroll
  for (int i = 0; i < 16; ++i) a[i] = (_Float16)p[(i >= 8 ? 16 : 0) + (i & 7)];
  return a;
}

// ---- B-fragment (32x16, KxN) from N-major f16 memory ([N][K], K contiguous).
// lane: n = lane&15, kb = (lane>>4)*8 ; elem i -> K = (i>=8?16:0)+kb+(i&7)
// -> two contiguous 8xf16 runs at +kb and +kb+16 : two b128 loads.
__device__ __forceinline__ v16h load_b_f16v(const _Float16* p0, int ld) {
  int lane = threadIdx.x & 31;
  int n = lane & 15, kb = (lane >> 4) * 8;
  const _Float16* p = p0 + (size_t)n * ld + kb;
  v8h lo = *(const v8h*)(p);
  v8h hi = *(const v8h*)(p + 16);
  return __builtin_shufflevector(lo, hi, 0, 1, 2, 3, 4, 5, 6, 7,
                                         8, 9, 10, 11, 12, 13, 14, 15);
}

// ---------------------------------------------------------------------------
// Prep kernels: one-time f16 conversion / re-layout of operands.
// ---------------------------------------------------------------------------
__global__ void cvt_f16_kernel(const float* __restrict__ src,
                               _Float16* __restrict__ dst, int n) {
  for (int i = blockIdx.x * blockDim.x + threadIdx.x; i < n;
       i += gridDim.x * blockDim.x)
    dst[i] = (_Float16)src[i];
}
// W[in][out] (f32) -> WT[out][in] (f16): B-fragments become K-contiguous.
__global__ void transpose_w_kernel(const float* __restrict__ W,
                                   _Float16* __restrict__ WT) {
  int idx = blockIdx.x * blockDim.x + threadIdx.x;   // o*768 + i
  if (idx >= HIDN * HIDN) return;
  int o = idx / HIDN, i = idx - o * HIDN;
  WT[idx] = (_Float16)W[(size_t)i * HIDN + o];
}
// emb[t][h*64+d] (f32) -> out[h][d][t] (f16): scatter-GEMM B becomes K-contig.
__global__ void transpose_vemb_kernel(const float* __restrict__ emb,
                                      _Float16* __restrict__ out, int T) {
  int idx = blockIdx.x * blockDim.x + threadIdx.x;   // h*(DH*T) + d*T + t
  if (idx >= NHEAD * DH * T) return;
  int t = idx % T, d = (idx / T) % DH, h = idx / (T * DH);
  out[idx] = (_Float16)emb[(size_t)t * HIDN + h * DH + d];
}

// ---------------------------------------------------------------------------
// Kernel 1: fused QKV projection.  One wave -> 16x64 tile (4 accumulators).
// q,k stored head-major [b,h,l,d]; v stored transposed [b,h,d,l] (packed
// b128 stores) so attention AV B-fragments are K-contiguous.
// grid = (12, 256, 3), block = 32.
// ---------------------------------------------------------------------------
__global__ __launch_bounds__(32)
void qkv_proj_kernel(const _Float16* __restrict__ xh,
                     const _Float16* __restrict__ WqT, const float* __restrict__ bq,
                     const _Float16* __restrict__ WkT, const float* __restrict__ bk,
                     const _Float16* __restrict__ WvT, const float* __restrict__ bv,
                     _Float16* __restrict__ qf, _Float16* __restrict__ kf,
                     _Float16* __restrict__ vT) {
  int nt = blockIdx.x, mt = blockIdx.y, which = blockIdx.z;
  const _Float16* WT  = (which == 0) ? WqT : (which == 1) ? WkT : WvT;
  const float*   bias = (which == 0) ? bq  : (which == 1) ? bk  : bv;

  int row0 = mt * 16, col0 = nt * 64;
  v8f acc[4];
#pragma unroll
  for (int t = 0; t < 4; ++t) acc[t] = (v8f){};

  for (int kc = 0; kc < HIDN; kc += 32) {
    v16h a = load_a_f16v(xh + (size_t)row0 * HIDN + kc, HIDN);
#pragma unroll
    for (int t = 0; t < 4; ++t) {
      v16h b = load_b_f16v(WT + (size_t)(col0 + t * 16) * HIDN + kc, HIDN);
      acc[t] = wmma16(a, b, acc[t]);
    }
  }
  int lane = threadIdx.x & 31;
  int n = lane & 15, mh = (lane >> 4) * 8;
  if (which == 2) {                    // V: transposed layout, packed stores
#pragma unroll
    for (int t = 0; t < 4; ++t) {
      int col = col0 + t * 16 + n;
      int h = col >> 6, d = col & 63;
      float bval = bias[col];
      v8h pack;
#pragma unroll
      for (int r = 0; r < 8; ++r) pack[r] = (_Float16)(acc[t][r] + bval);
      int row = row0 + mh;
      int b_ = row >> 9, l = row & (LSEQ - 1);
      *(v8h*)(vT + (((size_t)b_ * NHEAD + h) * DH + d) * LSEQ + l) = pack;
    }
  } else {                             // Q/K: head-major [b,h,l,d]
    _Float16* dst = (which == 0) ? qf : kf;
#pragma unroll
    for (int t = 0; t < 4; ++t) {
      int col = col0 + t * 16 + n;
      int h = col >> 6, d = col & 63;
      float bval = bias[col];
#pragma unroll
      for (int r = 0; r < 8; ++r) {
        int row = row0 + mh + r;
        int b_ = row >> 9, l = row & (LSEQ - 1);
        dst[(((size_t)b_ * NHEAD + h) * LSEQ + l) * DH + d] =
            (_Float16)(acc[t][r] + bval);
      }
    }
  }
}

// ---------------------------------------------------------------------------
// Kernel 2: per-head type-bias projection tables (f16 embeddings).
//   qhP[b,h,l,t] = sum_d q[b,h,l,d]*qhop_emb[t,h,d]   (+ qe/kh/ke variants)
// grid = (4 t-tiles, 32 l-tiles, 4*B*NH), block = 32.
// ---------------------------------------------------------------------------
__global__ __launch_bounds__(32)
void biasproj_kernel(const _Float16* __restrict__ qf, const _Float16* __restrict__ kf,
                     const _Float16* __restrict__ qhE, const _Float16* __restrict__ qeE,
                     const _Float16* __restrict__ khE, const _Float16* __restrict__ keE,
                     float* __restrict__ qhP, float* __restrict__ qeP,
                     float* __restrict__ khP, float* __restrict__ keP) {
  int nt = blockIdx.x, lt = blockIdx.y, z = blockIdx.z;
  int which = z / (BSZ * NHEAD);
  int bh    = z % (BSZ * NHEAD);
  int T = (which == 0 || which == 2) ? THOP : TEDGE;
  if (nt * 16 >= T) return;
  const _Float16* src = (which < 2) ? qf : kf;
  const _Float16* emb = (which == 0) ? qhE : (which == 1) ? qeE
                       : (which == 2) ? khE : keE;
  float* dstP = (which == 0) ? qhP : (which == 1) ? qeP
               : (which == 2) ? khP : keP;
  int h = bh % NHEAD;

  const _Float16* A0 = src + (size_t)bh * LSEQ * DH + (size_t)lt * 16 * DH;
  v8f acc = {};
  for (int kc = 0; kc < DH; kc += 32) {
    v16h a = load_a_f16v(A0 + kc, DH);
    // embh [T][768]: row t, cols h*64+d -> N-major, K contiguous
    v16h b = load_b_f16v(emb + (size_t)(nt * 16) * HIDN + h * DH + kc, HIDN);
    acc = wmma16(a, b, acc);
  }
  int lane = threadIdx.x & 31;
  int n = lane & 15, mh = (lane >> 4) * 8;
  float* dst = dstP + (size_t)bh * LSEQ * T + (size_t)(lt * 16) * T + nt * 16 + n;
#pragma unroll
  for (int r = 0; r < 8; ++r) dst[(size_t)(mh + r) * T] = acc[r];
}

// ---------------------------------------------------------------------------
// Kernel 3: attention. One wave owns a 16-query strip of one (b,h).
// grid = (32 q-tiles, 96 b*h), block = 32.  LDS = 54 KB / block.
// ---------------------------------------------------------------------------
__global__ __launch_bounds__(32)
void attn_kernel(const _Float16* __restrict__ qf, const _Float16* __restrict__ kf,
                 const _Float16* __restrict__ vT,
                 const float* __restrict__ qhP, const float* __restrict__ qeP,
                 const float* __restrict__ khP, const float* __restrict__ keP,
                 const int* __restrict__ hop, const int* __restrict__ edge,
                 const _Float16* __restrict__ vhT, const _Float16* __restrict__ veT,
                 _Float16* __restrict__ att_out) {
  alignas(16) __shared__ float    s_scores[16 * LSEQ];          // 32 KB
  alignas(16) __shared__ _Float16 s_att[16 * LSEQ];             // 16 KB
  alignas(16) __shared__ float    s_scat[16 * (THOP + TEDGE)];  //  6 KB

  int qt = blockIdx.x, bh = blockIdx.y;
  int b = bh / NHEAD, h = bh % NHEAD;
  int lane = threadIdx.x & 31;
  int n = lane & 15, mh = (lane >> 4) * 8;
  int row0 = qt * 16;

  const _Float16* qb  = qf + (size_t)bh * LSEQ * DH;
  const _Float16* kb  = kf + (size_t)bh * LSEQ * DH;
  const _Float16* vTb = vT + (size_t)bh * DH * LSEQ;

  // --- 1. raw scores q.k^T into LDS strip; q fragments hoisted ---
  v16h aq0 = load_a_f16v(qb + (size_t)row0 * DH + 0,  DH);
  v16h aq1 = load_a_f16v(qb + (size_t)row0 * DH + 32, DH);
  for (int kt = 0; kt < LSEQ / 16; ++kt) {
    v8f acc = {};
    v16h b0 = load_b_f16v(kb + (size_t)(kt * 16) * DH + 0,  DH);
    v16h b1 = load_b_f16v(kb + (size_t)(kt * 16) * DH + 32, DH);
    acc = wmma16(aq0, b0, acc);
    acc = wmma16(aq1, b1, acc);
#pragma unroll
    for (int r = 0; r < 8; ++r) s_scores[(mh + r) * LSEQ + kt * 16 + n] = acc[r];
  }
  __syncthreads();

  // --- 2. add gathered type biases, apply 1/sqrt(64) ---
  const float scale = 0.125f;
  const float* qhR = qhP + (size_t)bh * LSEQ * THOP;
  const float* qeR = qeP + (size_t)bh * LSEQ * TEDGE;
  const float* khR = khP + (size_t)bh * LSEQ * THOP;
  const float* keR = keP + (size_t)bh * LSEQ * TEDGE;
  const int* hopB = hop  + (size_t)b * LSEQ * LSEQ;
  const int* edgB = edge + (size_t)b * LSEQ * LSEQ;
  for (int e = lane; e < 16 * LSEQ; e += 32) {
    int m = e >> 9, j = e & (LSEQ - 1);
    int ig = row0 + m;
    int ht = hopB[(size_t)ig * LSEQ + j];
    int et = edgB[(size_t)ig * LSEQ + j];
    float val = s_scores[e]
              + qhR[(size_t)ig * THOP + ht] + qeR[(size_t)ig * TEDGE + et]
              + khR[(size_t)j  * THOP + ht] + keR[(size_t)j  * TEDGE + et];
    s_scores[e] = val * scale;
  }
  __syncthreads();

  // --- 3. row softmax (one lane per query row) ---
  if (lane < 16) {
    float mx = -1e30f;
    for (int j = 0; j < LSEQ; ++j) mx = fmaxf(mx, s_scores[lane * LSEQ + j]);
    float sum = 0.f;
    for (int j = 0; j < LSEQ; ++j) {
      float ev = __expf(s_scores[lane * LSEQ + j] - mx);
      s_scores[lane * LSEQ + j] = ev;
      sum += ev;
    }
    float inv = 1.f / sum;
    for (int j = 0; j < LSEQ; ++j) {
      float a = s_scores[lane * LSEQ + j] * inv;
      s_scores[lane * LSEQ + j] = a;
      s_att[lane * LSEQ + j] = (_Float16)a;
    }
  }
  for (int e = lane; e < 16 * (THOP + TEDGE); e += 32) s_scat[e] = 0.f;
  __syncthreads();

  // --- 4. scatter attention mass by type -> [16 x 96] (ds_add_f32) ---
  for (int e = lane; e < 16 * LSEQ; e += 32) {
    int m = e >> 9, j = e & (LSEQ - 1);
    int ig = row0 + m;
    int ht = hopB[(size_t)ig * LSEQ + j];
    int et = edgB[(size_t)ig * LSEQ + j];
    float a = s_scores[e];
    atomicAdd(&s_scat[m * (THOP + TEDGE) + ht], a);
    atomicAdd(&s_scat[m * (THOP + TEDGE) + THOP + et], a);
  }
  __syncthreads();

  // --- 5. out = att@V + scat_hop@vhopT + scat_edge@vedgeT (all WMMA) ---
  v8f acc[4];
#pragma unroll
  for (int t = 0; t < 4; ++t) acc[t] = (v8f){};
  for (int jc = 0; jc < LSEQ; jc += 32) {
    v16h a = load_a_f16v(s_att + jc, LSEQ);
#pragma unroll
    for (int t = 0; t < 4; ++t) {
      // vT [b,h,d,l]: B(K=j, N=d) is K-contiguous
      v16h bf = load_b_f16v(vTb + (size_t)(t * 16) * LSEQ + jc, LSEQ);
      acc[t] = wmma16(a, bf, acc[t]);
    }
  }
  {  // hop types: K = 32
    v16h a = load_a_f32(s_scat, THOP + TEDGE);
#pragma unroll
    for (int t = 0; t < 4; ++t) {
      v16h bf = load_b_f16v(vhT + ((size_t)h * DH + t * 16) * THOP, THOP);
      acc[t] = wmma16(a, bf, acc[t]);
    }
  }
  for (int kc = 0; kc < TEDGE; kc += 32) {  // edge types: K = 64
    v16h a = load_a_f32(s_scat + THOP + kc, THOP + TEDGE);
#pragma unroll
    for (int t = 0; t < 4; ++t) {
      v16h bf = load_b_f16v(veT + ((size_t)h * DH + t * 16) * TEDGE + kc, TEDGE);
      acc[t] = wmma16(a, bf, acc[t]);
    }
  }
  // store f16 [B, L, HID] for the output projection
#pragma unroll
  for (int t = 0; t < 4; ++t) {
#pragma unroll
    for (int r = 0; r < 8; ++r) {
      int l = row0 + mh + r;
      att_out[((size_t)b * LSEQ + l) * HIDN + h * DH + t * 16 + n] =
          (_Float16)acc[t][r];
    }
  }
}

// ---------------------------------------------------------------------------
// Kernel 4: output projection. One wave -> 16x64 tile. grid = (12, 256).
// ---------------------------------------------------------------------------
__global__ __launch_bounds__(32)
void outproj_kernel(const _Float16* __restrict__ att_out,
                    const _Float16* __restrict__ WoT, const float* __restrict__ bo,
                    float* __restrict__ out) {
  int nt = blockIdx.x, mt = blockIdx.y;
  int row0 = mt * 16, col0 = nt * 64;
  v8f acc[4];
#pragma unroll
  for (int t = 0; t < 4; ++t) acc[t] = (v8f){};
  for (int kc = 0; kc < HIDN; kc += 32) {
    v16h a = load_a_f16v(att_out + (size_t)row0 * HIDN + kc, HIDN);
#pragma unroll
    for (int t = 0; t < 4; ++t) {
      v16h b = load_b_f16v(WoT + (size_t)(col0 + t * 16) * HIDN + kc, HIDN);
      acc[t] = wmma16(a, b, acc[t]);
    }
  }
  int lane = threadIdx.x & 31;
  int n = lane & 15, mh = (lane >> 4) * 8;
#pragma unroll
  for (int t = 0; t < 4; ++t) {
    float bval = bo[col0 + t * 16 + n];
#pragma unroll
    for (int r = 0; r < 8; ++r)
      out[(size_t)(row0 + mh + r) * HIDN + col0 + t * 16 + n] = acc[t][r] + bval;
  }
}

// ---------------------------------------------------------------------------
extern "C" void kernel_launch(void* const* d_in, const int* in_sizes, int n_in,
                              void* d_out, int out_size, void* d_ws, size_t ws_size,
                              hipStream_t stream) {
  const float* x       = (const float*)d_in[0];
  const float* qh_emb  = (const float*)d_in[1];
  const float* qe_emb  = (const float*)d_in[2];
  const float* kh_emb  = (const float*)d_in[3];
  const float* ke_emb  = (const float*)d_in[4];
  const float* vh_emb  = (const float*)d_in[5];
  const float* ve_emb  = (const float*)d_in[6];
  const int*   hop     = (const int*)d_in[7];
  const int*   edge    = (const int*)d_in[8];
  const float* Wq      = (const float*)d_in[9];
  const float* bq      = (const float*)d_in[10];
  const float* Wk      = (const float*)d_in[11];
  const float* bk      = (const float*)d_in[12];
  const float* Wv      = (const float*)d_in[13];
  const float* bv      = (const float*)d_in[14];
  const float* Wo      = (const float*)d_in[15];
  const float* bo      = (const float*)d_in[16];

  // ---- workspace carving ----
  char* ws = (char*)d_ws;
  size_t off = 0;
  const size_t nX    = (size_t)BSZ * LSEQ * HIDN;          // 3.1M
  const size_t nBHLD = (size_t)BSZ * NHEAD * LSEQ * DH;    // 3.1M
  const size_t nBHL  = (size_t)BSZ * NHEAD * LSEQ;
  _Float16* xh  = (_Float16*)(ws + off); off += nX * 2;
  _Float16* qf  = (_Float16*)(ws + off); off += nBHLD * 2;
  _Float16* kf  = (_Float16*)(ws + off); off += nBHLD * 2;
  _Float16* vT  = (_Float16*)(ws + off); off += nBHLD * 2;
  _Float16* WqT = (_Float16*)(ws + off); off += (size_t)HIDN * HIDN * 2;
  _Float16* WkT = (_Float16*)(ws + off); off += (size_t)HIDN * HIDN * 2;
  _Float16* WvT = (_Float16*)(ws + off); off += (size_t)HIDN * HIDN * 2;
  _Float16* WoT = (_Float16*)(ws + off); off += (size_t)HIDN * HIDN * 2;
  _Float16* qhE = (_Float16*)(ws + off); off += (size_t)THOP  * HIDN * 2;
  _Float16* qeE = (_Float16*)(ws + off); off += (size_t)TEDGE * HIDN * 2;
  _Float16* khE = (_Float16*)(ws + off); off += (size_t)THOP  * HIDN * 2;
  _Float16* keE = (_Float16*)(ws + off); off += (size_t)TEDGE * HIDN * 2;
  _Float16* vhT = (_Float16*)(ws + off); off += (size_t)NHEAD * DH * THOP  * 2;
  _Float16* veT = (_Float16*)(ws + off); off += (size_t)NHEAD * DH * TEDGE * 2;
  float* qhP = (float*)(ws + off); off += nBHL * THOP  * 4;
  float* qeP = (float*)(ws + off); off += nBHL * TEDGE * 4;
  float* khP = (float*)(ws + off); off += nBHL * THOP  * 4;
  float* keP = (float*)(ws + off); off += nBHL * TEDGE * 4;
  _Float16* att_out = (_Float16*)(ws + off); off += nX * 2;

  dim3 blk(32), b256(256);

  // ---- 0. operand prep (f16 conversion / re-layout) ----
  cvt_f16_kernel<<<dim3(2048), b256, 0, stream>>>(x, xh, (int)nX);
  cvt_f16_kernel<<<dim3(96),  b256, 0, stream>>>(qh_emb, qhE, THOP * HIDN);
  cvt_f16_kernel<<<dim3(192), b256, 0, stream>>>(qe_emb, qeE, TEDGE * HIDN);
  cvt_f16_kernel<<<dim3(96),  b256, 0, stream>>>(kh_emb, khE, THOP * HIDN);
  cvt_f16_kernel<<<dim3(192), b256, 0, stream>>>(ke_emb, keE, TEDGE * HIDN);
  transpose_w_kernel<<<dim3((HIDN * HIDN + 255) / 256), b256, 0, stream>>>(Wq, WqT);
  transpose_w_kernel<<<dim3((HIDN * HIDN + 255) / 256), b256, 0, stream>>>(Wk, WkT);
  transpose_w_kernel<<<dim3((HIDN * HIDN + 255) / 256), b256, 0, stream>>>(Wv, WvT);
  transpose_w_kernel<<<dim3((HIDN * HIDN + 255) / 256), b256, 0, stream>>>(Wo, WoT);
  transpose_vemb_kernel<<<dim3((NHEAD * DH * THOP + 255) / 256), b256, 0, stream>>>(
      vh_emb, vhT, THOP);
  transpose_vemb_kernel<<<dim3((NHEAD * DH * TEDGE + 255) / 256), b256, 0, stream>>>(
      ve_emb, veT, TEDGE);

  // ---- 1. QKV projections ----
  qkv_proj_kernel<<<dim3(HIDN / 64, (BSZ * LSEQ) / 16, 3), blk, 0, stream>>>(
      xh, WqT, bq, WkT, bk, WvT, bv, qf, kf, vT);

  // ---- 2. type-bias projection tables ----
  biasproj_kernel<<<dim3(4, LSEQ / 16, 4 * BSZ * NHEAD), blk, 0, stream>>>(
      qf, kf, qhE, qeE, khE, keE, qhP, qeP, khP, keP);

  // ---- 3. attention + type scatter ----
  attn_kernel<<<dim3(LSEQ / 16, BSZ * NHEAD), blk, 0, stream>>>(
      qf, kf, vT, qhP, qeP, khP, keP, hop, edge, vhT, veT, att_out);

  // ---- 4. output projection -> d_out (f32) ----
  outproj_kernel<<<dim3(HIDN / 64, (BSZ * LSEQ) / 16), blk, 0, stream>>>(
      att_out, WoT, bo, (float*)d_out);
}
